// Gen_node_15573551415667
// MI455X (gfx1250) — compile-verified, matrix operand order checked by reading, not run on hardware
//
#include <hip/hip_runtime.h>
#include <hip/hip_bf16.h>

typedef __attribute__((ext_vector_type(16))) __bf16 v16bf;
typedef __attribute__((ext_vector_type(8)))  float  v8f;
typedef unsigned int u32x4 __attribute__((ext_vector_type(4)));
typedef int          i32x8 __attribute__((ext_vector_type(8)));
typedef int          i32x4 __attribute__((ext_vector_type(4)));

#define DIM 128
#define WPB 8               // waves per block
#define BLOCK (WPB * 32)
#define MAX_EBLOCKS 1024    // persistent grid caps
#define MAX_NBLOCKS 512

// ---------------------------------------------------------------------------
// zero a float buffer (grid-stride)
// ---------------------------------------------------------------------------
__global__ void zero_f32(float* __restrict__ p, size_t n) {
  size_t i = (size_t)blockIdx.x * blockDim.x + threadIdx.x;
  size_t s = (size_t)gridDim.x * blockDim.x;
  for (; i < n; i += s) p[i] = 0.0f;
}

// ---------------------------------------------------------------------------
// Shuffle a [K x 128] f32 weight matrix into bf16 B-fragment layout:
//   frag = (n_tile * (K/32) + k_step); per fragment 32 lanes x 16 bf16.
//   lane L holds column n = n_tile*16 + (L&15); element i holds
//   k = k_step*32 + ((i>=8)?16:0) + (L>=16?8:0) + (i&7)
// ---------------------------------------------------------------------------
__global__ void prep_weight(const float* __restrict__ W, __bf16* __restrict__ Wb, int K) {
  int ksteps = K >> 5;
  int total  = K * DIM;
  int tid = blockIdx.x * blockDim.x + threadIdx.x;
  if (tid >= total) return;
  int elem  = tid & 15;
  int lane  = (tid >> 4) & 31;
  int frag  = tid >> 9;                 // 512 elements per fragment
  int k_step = frag % ksteps;
  int n_tile = frag / ksteps;
  int half = lane >> 4;
  int kl = ((elem >= 8) ? 16 : 0) + half * 8 + (elem & 7);
  int k  = k_step * 32 + kl;
  int n  = n_tile * 16 + (lane & 15);
  Wb[tid] = (__bf16)W[k * DIM + n];
}

// ---------------------------------------------------------------------------
// Stage the fragment-shuffled weight panel (nelems bf16, contiguous) into LDS
// at offset 0, using the Tensor Data Mover (TENSORcnt path).
// D# built per CDNA5 ISA ch.8: 1-D tile of `nelems` 2-byte elements.
// ---------------------------------------------------------------------------
__device__ __forceinline__
void stage_weights(const __bf16* __restrict__ Wb, __bf16* smw, int nelems) {
#if __has_builtin(__builtin_amdgcn_tensor_load_to_lds)
  if (threadIdx.x < 32) {
    unsigned long long ga = (unsigned long long)(uintptr_t)Wb;
    unsigned ne = (unsigned)nelems;               // <= 49152, fits tile_dim0
    u32x4 g0;
    g0[0] = 1u;                                   // count=1, user descriptor
    g0[1] = 0u;                                   // lds_addr = 0 (dyn-LDS base)
    g0[2] = (unsigned)(ga & 0xFFFFFFFFu);         // global_addr[31:0]
    g0[3] = (unsigned)((ga >> 32) & 0x01FFFFFFu)  // global_addr[56:32]
          | 0x80000000u;                          // type = 2 ("image")
    i32x8 g1;
    g1[0] = 0x00010000;                           // data_size = 2 bytes
    g1[1] = (int)((ne & 0xFFFFu) << 16);          // tensor_dim0 lo16 @ b63:48
    g1[2] = (int)((ne >> 16) & 0xFFFFu);          // tensor_dim0 hi16 @ b79:64
    g1[3] = (int)((ne & 0xFFFFu) << 16);          // tile_dim0 @ b127:112
    g1[4] = 0;                                    // tile_dim1/2 unused (1-D)
    g1[5] = (int)ne;                              // tensor_dim0_stride lo32
    g1[6] = 0;
    g1[7] = 0;
    i32x4 g2; g2[0] = 0; g2[1] = 0; g2[2] = 0; g2[3] = 0;
    i32x4 g3 = g2;
    i32x8 g4; g4[0] = 0; g4[1] = 0; g4[2] = 0; g4[3] = 0;
              g4[4] = 0; g4[5] = 0; g4[6] = 0; g4[7] = 0;
    __builtin_amdgcn_tensor_load_to_lds(g0, g1, g2, g3, g4, 0);
#if __has_builtin(__builtin_amdgcn_s_wait_tensorcnt)
    __builtin_amdgcn_s_wait_tensorcnt(0);
#else
    asm volatile("s_wait_tensorcnt 0x0" ::: "memory");
#endif
  }
  __syncthreads();
#else
  const uint4* src = (const uint4*)Wb;
  uint4* dst = (uint4*)smw;
  int n16 = nelems >> 3;                          // 8 bf16 per 16B chunk
  for (int i = threadIdx.x; i < n16; i += BLOCK) dst[i] = src[i];
  __syncthreads();
#endif
}

// ---------------------------------------------------------------------------
// Edge MLP + scatter.  Persistent blocks; one wave per 16-edge tile iteration;
// weights resident in LDS (staged once per block via TDM).
//   A row m = x[src[m]] (k 0..127) ++ x[dst[m]] (k 128..255) [++ eprev[m] bf16]
//   e = relu(A @ We + be) -> eout (bf16);  atomicAdd(agg[dst], e * norm)
// ---------------------------------------------------------------------------
template <int KSEGS>   // 2 (layer 0) or 3 (layers 1,2)
__global__ __launch_bounds__(BLOCK)
void edge_kernel(const int* __restrict__ ei, const float* __restrict__ x,
                 const __bf16* __restrict__ eprev, const float* __restrict__ norm,
                 const __bf16* __restrict__ Wb, const float* __restrict__ bias,
                 __bf16* __restrict__ eout, float* __restrict__ agg, int E)
{
  extern __shared__ __bf16 smw[];
  constexpr int ksteps = KSEGS * 4;
  stage_weights(Wb, smw, ksteps * 32 * 128);      // K*128 bf16

  const int lane = threadIdx.x & 31;
  const int m    = lane & 15;
  const int half = lane >> 4;
  const int ncol = lane & 15;
  const int ntile = (E + 15) >> 4;
  const int tstride = gridDim.x * WPB;

  float bv[8];
#pragma unroll
  for (int t = 0; t < 8; ++t) bv[t] = bias[t * 16 + ncol];

  for (int tile = blockIdx.x * WPB + (threadIdx.x >> 5); tile < ntile; tile += tstride) {
    // Launder a zero LDS offset each iteration so the B-fragment ds_loads
    // cannot be hoisted out of the tile loop (which would spill to scratch).
    int smoff = 0;
    asm volatile("" : "+v"(smoff));
    const __bf16* wls = smw + smoff;

    const int edge0 = tile << 4;
    const int myedge = min(edge0 + m, E - 1);
    const int src  = ei[myedge];
    const int dstm = ei[E + myedge];
    const float*  rsrc = x + (size_t)src  * DIM;
    const float*  rdst = x + (size_t)dstm * DIM;
    const __bf16* rpre = (KSEGS == 3) ? (eprev + (size_t)myedge * DIM)
                                      : (const __bf16*)nullptr;

    v8f acc[8];
#pragma unroll
    for (int t = 0; t < 8; ++t)
#pragma unroll
      for (int j = 0; j < 8; ++j) acc[t][j] = 0.0f;

#pragma unroll
    for (int s = 0; s < ksteps; ++s) {
      v16bf a;
      const int off = (s & 3) * 32 + half * 8;    // compile-time per step
      if (s < 4) {
        const float* base = rsrc + off;
#pragma unroll
        for (int i = 0; i < 8; ++i) { a[i] = (__bf16)base[i]; a[i+8] = (__bf16)base[i+16]; }
      } else if (s < 8) {
        const float* base = rdst + off;
#pragma unroll
        for (int i = 0; i < 8; ++i) { a[i] = (__bf16)base[i]; a[i+8] = (__bf16)base[i+16]; }
      } else {
        const __bf16* base = rpre + off;
#pragma unroll
        for (int i = 0; i < 8; ++i) { a[i] = base[i]; a[i+8] = base[i+16]; }
      }
#pragma unroll
      for (int t = 0; t < 8; ++t) {
        const v16bf b = *(const v16bf*)(wls + (((t * ksteps + s) * 32 + lane) << 4));
        acc[t] = __builtin_amdgcn_wmma_f32_16x16x32_bf16(
            false, a, false, b, (short)0, acc[t], false, false);
      }
    }

    // epilogue: bias + relu, store e (bf16), scatter e*norm (f32 atomics)
    float nrm[8];
    int   dstv[8];
    bool  okr[8];
#pragma unroll
    for (int r = 0; r < 8; ++r) {
      int erow = edge0 + r + half * 8;
      okr[r]  = erow < E;
      int ec  = okr[r] ? erow : (E - 1);
      nrm[r]  = norm[ec];
      dstv[r] = ei[E + ec];
    }
#pragma unroll
    for (int t = 0; t < 8; ++t) {
      const int n = t * 16 + ncol;
#pragma unroll
      for (int r = 0; r < 8; ++r) {
        if (!okr[r]) continue;
        const int erow = edge0 + r + half * 8;
        float v = acc[t][r] + bv[t];
        v = v > 0.0f ? v : 0.0f;
        eout[(size_t)erow * DIM + n] = (__bf16)v;
        atomicAdd(&agg[(size_t)dstv[r] * DIM + n], v * nrm[r]);
      }
    }
  }
}

// ---------------------------------------------------------------------------
// Node MLP.  Persistent blocks; one wave per 16-node tile iteration.
//   A row = x[i] ++ agg[i]  (K = 256);  xout = relu(A @ Wn + bn)
// ---------------------------------------------------------------------------
__global__ __launch_bounds__(BLOCK)
void node_kernel(const float* __restrict__ x, const float* __restrict__ agg,
                 const __bf16* __restrict__ Wb, const float* __restrict__ bias,
                 float* __restrict__ xout, int N)
{
  extern __shared__ __bf16 smw[];
  constexpr int ksteps = 8;                       // K = 256
  stage_weights(Wb, smw, ksteps * 32 * 128);

  const int lane = threadIdx.x & 31;
  const int m    = lane & 15;
  const int half = lane >> 4;
  const int ncol = lane & 15;
  const int ntile = (N + 15) >> 4;
  const int tstride = gridDim.x * WPB;

  float bv[8];
#pragma unroll
  for (int t = 0; t < 8; ++t) bv[t] = bias[t * 16 + ncol];

  for (int tile = blockIdx.x * WPB + (threadIdx.x >> 5); tile < ntile; tile += tstride) {
    int smoff = 0;
    asm volatile("" : "+v"(smoff));
    const __bf16* wls = smw + smoff;

    const int node0 = tile << 4;
    const int mynode = min(node0 + m, N - 1);
    const float* rx = x   + (size_t)mynode * DIM;
    const float* ra = agg + (size_t)mynode * DIM;

    v8f acc[8];
#pragma unroll
    for (int t = 0; t < 8; ++t)
#pragma unroll
      for (int j = 0; j < 8; ++j) acc[t][j] = 0.0f;

#pragma unroll
    for (int s = 0; s < ksteps; ++s) {
      v16bf a;
      const float* base = ((s < 4) ? rx : ra) + (s & 3) * 32 + half * 8;
#pragma unroll
      for (int i = 0; i < 8; ++i) { a[i] = (__bf16)base[i]; a[i+8] = (__bf16)base[i+16]; }
#pragma unroll
      for (int t = 0; t < 8; ++t) {
        const v16bf b = *(const v16bf*)(wls + (((t * ksteps + s) * 32 + lane) << 4));
        acc[t] = __builtin_amdgcn_wmma_f32_16x16x32_bf16(
            false, a, false, b, (short)0, acc[t], false, false);
      }
    }

#pragma unroll
    for (int t = 0; t < 8; ++t) {
      const int n = t * 16 + ncol;
#pragma unroll
      for (int r = 0; r < 8; ++r) {
        const int nrow = node0 + r + half * 8;
        if (nrow >= N) continue;
        float v = acc[t][r] + bv[t];
        v = v > 0.0f ? v : 0.0f;
        xout[(size_t)nrow * DIM + n] = v;
      }
    }
  }
}

// ---------------------------------------------------------------------------
// launch
// ---------------------------------------------------------------------------
extern "C" void kernel_launch(void* const* d_in, const int* in_sizes, int n_in,
                              void* d_out, int out_size, void* d_ws, size_t ws_size,
                              hipStream_t stream)
{
  const int*   ei   = (const int*)d_in[0];
  const float* z    = (const float*)d_in[1];
  const float* norm = (const float*)d_in[2];
  const float* We0  = (const float*)d_in[3];  const float* be0 = (const float*)d_in[4];
  const float* Wn0  = (const float*)d_in[5];  const float* bn0 = (const float*)d_in[6];
  const float* We1  = (const float*)d_in[7];  const float* be1 = (const float*)d_in[8];
  const float* Wn1  = (const float*)d_in[9];  const float* bn1 = (const float*)d_in[10];
  const float* We2  = (const float*)d_in[11]; const float* be2 = (const float*)d_in[12];
  const float* Wn2  = (const float*)d_in[13]; const float* bn2 = (const float*)d_in[14];

  const int E = in_sizes[2];
  const int N = in_sizes[1] / DIM;

  const size_t eElems = (size_t)E * DIM;
  const size_t xElems = (size_t)N * DIM;

  // workspace carve-up (all section sizes are multiples of 32 bytes)
  __bf16* eA  = (__bf16*)d_ws;
  __bf16* eB  = eA + eElems;
  float*  xA  = (float*)(eB + eElems);
  float*  xB  = xA + xElems;
  float*  agg = xB + xElems;
  __bf16* wbe0 = (__bf16*)(agg + xElems);
  __bf16* wbn0 = wbe0 + 256 * DIM;
  __bf16* wbe1 = wbn0 + 256 * DIM;
  __bf16* wbn1 = wbe1 + 384 * DIM;
  __bf16* wbe2 = wbn1 + 256 * DIM;
  __bf16* wbn2 = wbe2 + 384 * DIM;

  // weight shuffles (tiny)
  prep_weight<<<(256 * DIM + 255) / 256, 256, 0, stream>>>(We0, wbe0, 256);
  prep_weight<<<(256 * DIM + 255) / 256, 256, 0, stream>>>(Wn0, wbn0, 256);
  prep_weight<<<(384 * DIM + 255) / 256, 256, 0, stream>>>(We1, wbe1, 384);
  prep_weight<<<(256 * DIM + 255) / 256, 256, 0, stream>>>(Wn1, wbn1, 256);
  prep_weight<<<(384 * DIM + 255) / 256, 256, 0, stream>>>(We2, wbe2, 384);
  prep_weight<<<(256 * DIM + 255) / 256, 256, 0, stream>>>(Wn2, wbn2, 256);

  const int etiles  = (E + 15) / 16;
  const int ntiles  = (N + 15) / 16;
  int eblocks = (etiles + WPB - 1) / WPB; if (eblocks > MAX_EBLOCKS) eblocks = MAX_EBLOCKS;
  int nblocks = (ntiles + WPB - 1) / WPB; if (nblocks > MAX_NBLOCKS) nblocks = MAX_NBLOCKS;

  const size_t lds2 = 256 * DIM * sizeof(__bf16); // 64 KB (K=256)
  const size_t lds3 = 384 * DIM * sizeof(__bf16); // 96 KB (K=384)

  // layer 0
  zero_f32<<<2048, 256, 0, stream>>>(agg, xElems);
  edge_kernel<2><<<eblocks, BLOCK, lds2, stream>>>(ei, z, nullptr, norm, wbe0, be0, eA, agg, E);
  node_kernel<<<nblocks, BLOCK, lds2, stream>>>(z, agg, wbn0, bn0, xA, N);

  // layer 1
  zero_f32<<<2048, 256, 0, stream>>>(agg, xElems);
  edge_kernel<3><<<eblocks, BLOCK, lds3, stream>>>(ei, xA, eA, norm, wbe1, be1, eB, agg, E);
  node_kernel<<<nblocks, BLOCK, lds2, stream>>>(xA, agg, wbn1, bn1, xB, N);

  // layer 2
  zero_f32<<<2048, 256, 0, stream>>>(agg, xElems);
  edge_kernel<3><<<eblocks, BLOCK, lds3, stream>>>(ei, xB, eB, norm, wbe2, be2, eA, agg, E);
  node_kernel<<<nblocks, BLOCK, lds2, stream>>>(xB, agg, wbn2, bn2, (float*)d_out, N);
}
